// GCN_LSTM_13116830122541
// MI455X (gfx1250) — matrix operand order repaired and checked by reading, not compiled, not run on hardware
//
#include <hip/hip_runtime.h>
#include <cstdint>
#include <cstddef>

#define N_NODES 10000
#define M_PAD   10016             /* N_NODES rounded up to 32 */
#define N_EDGES 160000
#define F_NODE  128
#define C_GCN   64
#define S_SEQ   32
#define T_SEQ   96
#define H_LSTM  128
#define G4H     (4 * H_LSTM)      /* 512 */
#define IN0     (S_SEQ + C_GCN)   /* 96  */

typedef __attribute__((ext_vector_type(16))) __bf16 v16bf;
typedef __attribute__((ext_vector_type(8)))  float  v8f;

union Frag32B { uint4 q[2]; v16bf v; };

__device__ __forceinline__ unsigned short f2bf(float f) {
  unsigned int u = __float_as_uint(f);
  u += 0x7FFFu + ((u >> 16) & 1u);          // round to nearest even
  return (unsigned short)(u >> 16);
}

__device__ __forceinline__ float sigm(float x) {
  return 1.0f / (1.0f + __expf(-x));
}

// ---------------------------------------------------------------------------
// Generic init: fill n floats with value v (also used to zero bf16 buffers).
// ---------------------------------------------------------------------------
__global__ void init_f32(float* __restrict__ p, long long n, float v) {
  long long i = (long long)blockIdx.x * blockDim.x + threadIdx.x;
  if (i < n) p[i] = v;
}

// f32 -> bf16 elementwise
__global__ void cvt_bf16(const float* __restrict__ src,
                         unsigned short* __restrict__ dst, long long n) {
  long long i = (long long)blockIdx.x * blockDim.x + threadIdx.x;
  if (i < n) dst[i] = f2bf(src[i]);
}

// Zero the 16 pad rows of each time-step block of `comb`.
__global__ void zero_comb_pad(unsigned short* __restrict__ comb) {
  int i = blockIdx.x * blockDim.x + threadIdx.x;
  const int PAD = M_PAD - N_NODES;
  if (i >= T_SEQ * PAD * IN0) return;
  int c = i % IN0;
  int r = i / IN0;
  int pr = r % PAD;
  int t  = r / PAD;
  comb[((size_t)t * M_PAD + N_NODES + pr) * IN0 + c] = 0;
}

// ---------------------------------------------------------------------------
// Pack a K x N fp32 weight matrix into per-lane bf16 WMMA B-fragments.
// transposed==1: B[k][n] = src[n*ldsrc + k]  (for torch-style [out,in] weights)
// transposed==0: B[k][n] = src[k*ldsrc + n]
// Fragment order must match gemm_pass: (((tn*ksteps)+ks)*32 + lane)*16.
// B 32x16 bf16 layout: lanes 0-15 hold K=0..7,16..23 (col = lane), lanes 16-31 K=8..15,24..31.
// ---------------------------------------------------------------------------
__global__ void pack_b_bf16(const float* __restrict__ src,
                            unsigned short* __restrict__ dst,
                            int K, int N, int ldsrc, int transposed) {
  int tid = blockIdx.x * blockDim.x + threadIdx.x;
  int ksteps = K >> 5;
  int total = (N >> 4) * ksteps * 32;
  if (tid >= total) return;
  int lane = tid & 31;
  int rem  = tid >> 5;
  int ks   = rem % ksteps;
  int tn   = rem / ksteps;
  int n    = (tn << 4) + (lane & 15);
  int kb   = (lane < 16) ? 0 : 8;
  unsigned short* out = dst + (size_t)tid * 16;
#pragma unroll
  for (int e = 0; e < 16; ++e) {
    int k = (ks << 5) + kb + e + (e >= 8 ? 8 : 0);
    float v = transposed ? src[(size_t)n * ldsrc + k]
                         : src[(size_t)k * ldsrc + n];
    out[e] = f2bf(v);
  }
}

// ---------------------------------------------------------------------------
// One GEMM accumulation pass over K for a 32x64 register tile.
// Branch-free inner loop: 12 b128 loads + prefetch + 8 WMMAs per K-step.
// ---------------------------------------------------------------------------
__device__ __forceinline__ void gemm_pass(const unsigned short* __restrict__ A,
                                          const unsigned short* __restrict__ Bp,
                                          int K, int lda,
                                          int m0, int tn0, int lane,
                                          v8f acc[8]) {
  const int lr = lane & 15;
  const int hi = lane >> 4;
  const int ksteps = K >> 5;
  // A-fragment base: row m0+lr, k-offset hi*8 (ISA 16-bit A layout)
  const unsigned short* arow0 = A + (size_t)(m0 + lr) * lda + hi * 8;
  const unsigned short* arow1 = arow0 + (size_t)16 * lda;

  for (int ks = 0; ks < ksteps; ++ks) {
    Frag32B fa0, fa1;
    fa0.q[0] = *(const uint4*)(arow0 + (ks << 5));        // K = kb+0..7
    fa0.q[1] = *(const uint4*)(arow0 + (ks << 5) + 16);   // K = kb+16..23
    fa1.q[0] = *(const uint4*)(arow1 + (ks << 5));
    fa1.q[1] = *(const uint4*)(arow1 + (ks << 5) + 16);
    __builtin_prefetch(arow0 + ((ks + 1) << 5), 0, 3);    // global_prefetch next A
#pragma unroll
    for (int j = 0; j < 4; ++j) {
      Frag32B fb;
      const unsigned short* bp =
          Bp + ((((size_t)(tn0 + j)) * ksteps + ks) * 32 + lane) * 16;
      fb.q[0] = ((const uint4*)bp)[0];
      fb.q[1] = ((const uint4*)bp)[1];
      acc[j] = __builtin_amdgcn_wmma_f32_16x16x32_bf16(
          false, fa0.v, false, fb.v, (short)0, acc[j], false, false);
      acc[4 + j] = __builtin_amdgcn_wmma_f32_16x16x32_bf16(
          false, fa1.v, false, fb.v, (short)0, acc[4 + j], false, false);
    }
  }
}

// ---------------------------------------------------------------------------
// Fused dual GEMM: C[M_PAD x N] = A0[M_PAD x K0] * B0 + A1[M_PAD x K1] * B1
// (second term skipped when K1 == 0). A row-major bf16, Bp pre-packed bf16
// fragments, C fp32 row-major (ldc == N). One wave per 32x64 output tile;
// M_PAD % 32 == 0 so the kernel is branch-free and EXEC stays all-ones.
// Used as gates = x_t @ Wih^T + h @ Whh^T in a single store (no C reload).
// ---------------------------------------------------------------------------
__global__ void __launch_bounds__(32)
wmma_gemm_dual(const unsigned short* __restrict__ A0,
               const unsigned short* __restrict__ Bp0, int K0, int lda0,
               const unsigned short* __restrict__ A1,
               const unsigned short* __restrict__ Bp1, int K1, int lda1,
               float* __restrict__ C, int N) {
  const int lane = threadIdx.x;
  const int lr = lane & 15;
  const int hi = lane >> 4;
  const int m0 = blockIdx.y << 5;          // 32-row tile base
  const int tn0 = blockIdx.x << 2;         // base 16-col tile of 64-col strip

  v8f acc[8];
  {
    v8f z = {};
#pragma unroll
    for (int j = 0; j < 8; ++j) acc[j] = z;
  }

  gemm_pass(A0, Bp0, K0, lda0, m0, tn0, lane, acc);
  gemm_pass(A1, Bp1, K1, lda1, m0, tn0, lane, acc);   // 0 iterations if K1==0

#pragma unroll
  for (int j = 0; j < 4; ++j) {
    int col = ((tn0 + j) << 4) + lr;
#pragma unroll
    for (int r = 0; r < 8; ++r) {
      C[(size_t)(m0 + r + hi * 8) * N + col] = acc[j][r];
      C[(size_t)(m0 + 16 + r + hi * 8) * N + col] = acc[4 + j][r];
    }
  }
}

// ---------------------------------------------------------------------------
// GCN pieces
// ---------------------------------------------------------------------------
__global__ void deg_accum(const int* __restrict__ ei, float* __restrict__ deg) {
  int e = blockIdx.x * blockDim.x + threadIdx.x;
  if (e < N_EDGES) atomicAdd(&deg[ei[N_EDGES + e]], 1.0f);
}

__global__ void deg_to_dinv(float* __restrict__ deg) {
  int n = blockIdx.x * blockDim.x + threadIdx.x;
  if (n < N_NODES) {
    float d = deg[n];
    deg[n] = (d > 0.0f) ? rsqrtf(d) : 0.0f;
  }
}

// self-loop term: out[n][c] = xw[n][c] * dinv[n]^2
__global__ void gcn_self(const float* __restrict__ xw,
                         const float* __restrict__ dinv,
                         float* __restrict__ out) {
  int i = blockIdx.x * blockDim.x + threadIdx.x;
  if (i < N_NODES * C_GCN) {
    float di = dinv[i / C_GCN];
    out[i] = xw[i] * di * di;
  }
}

// edge messages: out[col][c] += xw[row][c] * dinv[row]*dinv[col]
__global__ void gcn_scatter(const int* __restrict__ ei,
                            const float* __restrict__ xw,
                            const float* __restrict__ dinv,
                            float* __restrict__ out) {
  int i = blockIdx.x * blockDim.x + threadIdx.x;
  if (i >= N_EDGES * C_GCN) return;
  int e = i / C_GCN, c = i % C_GCN;
  int row = ei[e], col = ei[N_EDGES + e];
  float norm = dinv[row] * dinv[col];
  atomicAdd(&out[(size_t)col * C_GCN + c], xw[(size_t)row * C_GCN + c] * norm);
}

// ---------------------------------------------------------------------------
// Build time-major bf16 LSTM-0 input (rows padded to M_PAD per step):
// comb[(t*M_PAD + n)*IN0 + c] = c < 32 ? seq[n][t][c] : gcn_out[n][c-32]+gcn_b
// ---------------------------------------------------------------------------
__global__ void build_combined(const float* __restrict__ seq,
                               const float* __restrict__ gout,
                               const float* __restrict__ gb,
                               unsigned short* __restrict__ comb) {
  long long i = (long long)blockIdx.x * blockDim.x + threadIdx.x;
  if (i >= (long long)T_SEQ * N_NODES * IN0) return;
  int c = (int)(i % IN0);
  long long r = i / IN0;
  int n = (int)(r % N_NODES);
  int t = (int)(r / N_NODES);
  float v = (c < S_SEQ)
      ? seq[(size_t)n * T_SEQ * S_SEQ + (size_t)t * S_SEQ + c]
      : gout[(size_t)n * C_GCN + (c - S_SEQ)] + gb[c - S_SEQ];
  comb[((size_t)t * M_PAD + n) * IN0 + c] = f2bf(v);
}

// ---------------------------------------------------------------------------
// LSTM cell (torch gate order i,f,g,o): reads gates[M_PAD x 512] (first
// N_NODES rows), updates h,c, writes the bf16 copy of h for the next GEMM.
// ---------------------------------------------------------------------------
__global__ void lstm_cell(const float* __restrict__ gates,
                          const float* __restrict__ b_ih,
                          const float* __restrict__ b_hh,
                          float* __restrict__ h, float* __restrict__ c,
                          unsigned short* __restrict__ hb) {
  int i = blockIdx.x * blockDim.x + threadIdx.x;
  if (i >= N_NODES * H_LSTM) return;
  int n = i / H_LSTM, j = i % H_LSTM;
  const float* g = gates + (size_t)n * G4H;
  float gi = g[j]               + b_ih[j]               + b_hh[j];
  float gf = g[H_LSTM + j]      + b_ih[H_LSTM + j]      + b_hh[H_LSTM + j];
  float gg = g[2 * H_LSTM + j]  + b_ih[2 * H_LSTM + j]  + b_hh[2 * H_LSTM + j];
  float go = g[3 * H_LSTM + j]  + b_ih[3 * H_LSTM + j]  + b_hh[3 * H_LSTM + j];
  float cn = sigm(gf) * c[i] + sigm(gi) * tanhf(gg);
  float hn = sigm(go) * tanhf(cn);
  c[i] = cn;
  h[i] = hn;
  hb[i] = f2bf(hn);
}

// Final head: out[n] = h1_last[n] . fc_w + fc_b
__global__ void fc_head(const float* __restrict__ h,
                        const float* __restrict__ fw,
                        const float* __restrict__ fb,
                        float* __restrict__ out) {
  int n = blockIdx.x * blockDim.x + threadIdx.x;
  if (n >= N_NODES) return;
  float s = 0.0f;
#pragma unroll 4
  for (int j = 0; j < H_LSTM; ++j) s += h[(size_t)n * H_LSTM + j] * fw[j];
  out[n] = s + fb[0];
}

// ---------------------------------------------------------------------------
// Host-side orchestration
// ---------------------------------------------------------------------------
static inline char* carve(char*& p, size_t bytes) {
  char* r = p;
  p += (bytes + 255) & ~(size_t)255;
  return r;
}

extern "C" void kernel_launch(void* const* d_in, const int* in_sizes, int n_in,
                              void* d_out, int out_size, void* d_ws, size_t ws_size,
                              hipStream_t stream) {
  (void)in_sizes; (void)n_in; (void)out_size; (void)ws_size;
  const float* seq   = (const float*)d_in[0];
  const int*   ei    = (const int*)  d_in[1];   // edge_index [2, E]
  /* d_in[2] edge_attr, d_in[4] node_indices: unused by the reference math */
  const float* nf    = (const float*)d_in[3];
  const float* gcn_w = (const float*)d_in[5];
  const float* gcn_b = (const float*)d_in[6];
  const float* w_ih0 = (const float*)d_in[7];
  const float* w_hh0 = (const float*)d_in[8];
  const float* b_ih0 = (const float*)d_in[9];
  const float* b_hh0 = (const float*)d_in[10];
  const float* w_ih1 = (const float*)d_in[11];
  const float* w_hh1 = (const float*)d_in[12];
  const float* b_ih1 = (const float*)d_in[13];
  const float* b_hh1 = (const float*)d_in[14];
  const float* fc_w  = (const float*)d_in[15];
  const float* fc_b  = (const float*)d_in[16];
  float* out = (float*)d_out;

  char* p = (char*)d_ws;
  float* xw    = (float*)carve(p, (size_t)M_PAD * C_GCN * 4);
  float* dinv  = (float*)carve(p, (size_t)N_NODES * 4);
  float* gout  = (float*)carve(p, (size_t)N_NODES * C_GCN * 4);
  unsigned short* nf16  = (unsigned short*)carve(p, (size_t)M_PAD * F_NODE * 2);
  unsigned short* gwp   = (unsigned short*)carve(p, (size_t)F_NODE * C_GCN * 2);
  unsigned short* wih0p = (unsigned short*)carve(p, (size_t)IN0 * G4H * 2);
  unsigned short* whh0p = (unsigned short*)carve(p, (size_t)H_LSTM * G4H * 2);
  unsigned short* wih1p = (unsigned short*)carve(p, (size_t)H_LSTM * G4H * 2);
  unsigned short* whh1p = (unsigned short*)carve(p, (size_t)H_LSTM * G4H * 2);
  float* gates = (float*)carve(p, (size_t)M_PAD * G4H * 4);
  float* h0 = (float*)carve(p, (size_t)N_NODES * H_LSTM * 4);
  float* c0 = (float*)carve(p, (size_t)N_NODES * H_LSTM * 4);
  float* h1 = (float*)carve(p, (size_t)N_NODES * H_LSTM * 4);
  float* c1 = (float*)carve(p, (size_t)N_NODES * H_LSTM * 4);
  unsigned short* h0b = (unsigned short*)carve(p, (size_t)M_PAD * H_LSTM * 2);
  unsigned short* h1b = (unsigned short*)carve(p, (size_t)M_PAD * H_LSTM * 2);
  unsigned short* comb = (unsigned short*)carve(p, (size_t)T_SEQ * M_PAD * IN0 * 2);

  const int TB = 256;
  auto blk = [](long long n, int tb) { return (unsigned)((n + tb - 1) / tb); };
  const unsigned mblocks = M_PAD / 32;

  // ---- one-time (per call) conversions / weight packing / pad zeroing ----
  init_f32<<<blk((long long)M_PAD * F_NODE / 2, TB), TB, 0, stream>>>(
      (float*)nf16, (long long)M_PAD * F_NODE / 2, 0.0f);   // zero incl. pad rows
  cvt_bf16<<<blk((long long)N_NODES * F_NODE, TB), TB, 0, stream>>>(
      nf, nf16, (long long)N_NODES * F_NODE);
  zero_comb_pad<<<blk((long long)T_SEQ * (M_PAD - N_NODES) * IN0, TB), TB, 0,
                  stream>>>(comb);
  pack_b_bf16<<<blk((C_GCN/16)*(F_NODE/32)*32, TB), TB, 0, stream>>>(
      gcn_w, gwp, F_NODE, C_GCN, C_GCN, 0);                 // B = gcn_w (K x N)
  pack_b_bf16<<<blk((G4H/16)*(IN0/32)*32, TB), TB, 0, stream>>>(
      w_ih0, wih0p, IN0, G4H, IN0, 1);                      // B = w_ih0^T
  pack_b_bf16<<<blk((G4H/16)*(H_LSTM/32)*32, TB), TB, 0, stream>>>(
      w_hh0, whh0p, H_LSTM, G4H, H_LSTM, 1);
  pack_b_bf16<<<blk((G4H/16)*(H_LSTM/32)*32, TB), TB, 0, stream>>>(
      w_ih1, wih1p, H_LSTM, G4H, H_LSTM, 1);
  pack_b_bf16<<<blk((G4H/16)*(H_LSTM/32)*32, TB), TB, 0, stream>>>(
      w_hh1, whh1p, H_LSTM, G4H, H_LSTM, 1);

  // ---- GCN ----
  init_f32<<<blk(N_NODES, TB), TB, 0, stream>>>(dinv, N_NODES, 1.0f); // self loop
  deg_accum<<<blk(N_EDGES, TB), TB, 0, stream>>>(ei, dinv);
  deg_to_dinv<<<blk(N_NODES, TB), TB, 0, stream>>>(dinv);
  // XW = node_features @ gcn_w  (single-pass WMMA; K1 = 0)
  wmma_gemm_dual<<<dim3(C_GCN / 64, mblocks), dim3(32), 0, stream>>>(
      nf16, gwp, F_NODE, F_NODE, nf16, gwp, 0, F_NODE, xw, C_GCN);
  gcn_self<<<blk((long long)N_NODES * C_GCN, TB), TB, 0, stream>>>(xw, dinv, gout);
  gcn_scatter<<<blk((long long)N_EDGES * C_GCN, TB), TB, 0, stream>>>(
      ei, xw, dinv, gout);

  // ---- LSTM input tensor (time-major, bf16, padded rows) ----
  build_combined<<<blk((long long)T_SEQ * N_NODES * IN0, TB), TB, 0, stream>>>(
      seq, gout, gcn_b, comb);

  // ---- init recurrent state (pad rows of h0b/h1b stay zero all call) ----
  init_f32<<<blk((long long)N_NODES * H_LSTM, TB), TB, 0, stream>>>(h0, (long long)N_NODES * H_LSTM, 0.0f);
  init_f32<<<blk((long long)N_NODES * H_LSTM, TB), TB, 0, stream>>>(c0, (long long)N_NODES * H_LSTM, 0.0f);
  init_f32<<<blk((long long)N_NODES * H_LSTM, TB), TB, 0, stream>>>(h1, (long long)N_NODES * H_LSTM, 0.0f);
  init_f32<<<blk((long long)N_NODES * H_LSTM, TB), TB, 0, stream>>>(c1, (long long)N_NODES * H_LSTM, 0.0f);
  init_f32<<<blk((long long)M_PAD * H_LSTM / 2, TB), TB, 0, stream>>>(
      (float*)h0b, (long long)M_PAD * H_LSTM / 2, 0.0f);
  init_f32<<<blk((long long)M_PAD * H_LSTM / 2, TB), TB, 0, stream>>>(
      (float*)h1b, (long long)M_PAD * H_LSTM / 2, 0.0f);

  // ---- both LSTM layers in lockstep over time; per step, each layer is ONE
  //      fused dual GEMM (input + recurrent K-ranges) + one cell kernel.
  //      All per-step state is L2-resident on MI455X (192 MB L2). ----
  const dim3 ggrid(G4H / 64, mblocks), gblk(32);
  for (int t = 0; t < T_SEQ; ++t) {
    const unsigned short* xt = comb + (size_t)t * M_PAD * IN0;
    // layer 0: gates = x_t @ Wih0^T + h0 @ Whh0^T
    wmma_gemm_dual<<<ggrid, gblk, 0, stream>>>(
        xt, wih0p, IN0, IN0, h0b, whh0p, H_LSTM, H_LSTM, gates, G4H);
    lstm_cell<<<blk((long long)N_NODES * H_LSTM, TB), TB, 0, stream>>>(
        gates, b_ih0, b_hh0, h0, c0, h0b);
    // layer 1: gates = h0_t @ Wih1^T + h1 @ Whh1^T
    wmma_gemm_dual<<<ggrid, gblk, 0, stream>>>(
        h0b, wih1p, H_LSTM, H_LSTM, h1b, whh1p, H_LSTM, H_LSTM, gates, G4H);
    lstm_cell<<<blk((long long)N_NODES * H_LSTM, TB), TB, 0, stream>>>(
        gates, b_ih1, b_hh1, h1, c1, h1b);
  }

  // ---- head ----
  fc_head<<<blk(N_NODES, TB), TB, 0, stream>>>(h1, fc_w, fc_b, out);
}
